// BiLSTM_CRF_37958920962242
// MI455X (gfx1250) — matrix operand order repaired and checked by reading, not compiled, not run on hardware
//
#include <hip/hip_runtime.h>
#include <hip/hip_bf16.h>

#define B_   128
#define S_   512
#define C_   16
#define E_   100
#define CE_  30
#define NF_  25
#define H_   128
#define T_   17
#define KPAD 192   // 175 padded to 6*32
#define G4H  512   // 4*H
#define MB_  8     // batch blocks per direction (16 rows each)

typedef __attribute__((ext_vector_type(16))) __bf16 v16bf;
typedef __attribute__((ext_vector_type(8)))  float  v8f;

union FragB16 { v16bf v; unsigned int d[8]; unsigned short u[16]; };

__device__ __forceinline__ unsigned short f2bf(float f) {
  unsigned int x = __float_as_uint(f);
  return (unsigned short)((x + 0x7FFFu + ((x >> 16) & 1u)) >> 16);
}
__device__ __forceinline__ float sigf(float x) { return 1.f / (1.f + __expf(-x)); }

// ---------------- zero output ----------------
__global__ void k_zero(float* out) { if (threadIdx.x == 0) out[0] = 0.f; }

// ---------------- weight prep: fp32 -> bf16 (pad K of W_ih to 192) ----------------
__global__ void k_prep(const float* Wihf, const float* Wihb,
                       const float* Whhf, const float* Whhb,
                       unsigned short* WihAll, unsigned short* WhhAll) {
  int i = blockIdx.x * blockDim.x + threadIdx.x;
  const int tot1 = 1024 * KPAD;           // [dir*512+g][192]
  const int tot2 = 2 * G4H * H_;          // [dir][512][128]
  if (i < tot1) {
    int n = i / KPAD, k = i % KPAD;
    float v = 0.f;
    if (k < 175) v = (n < 512) ? Wihf[n * 175 + k] : Wihb[(n - 512) * 175 + k];
    WihAll[i] = f2bf(v);
  }
  if (i < tot2) {
    int dir = i / (G4H * H_), r = i % (G4H * H_);
    WhhAll[i] = f2bf(dir ? Whhb[r] : Whhf[r]);
  }
}

// ---------------- embeddings + char convs -> feats bf16 [B*S][192] ----------------
__global__ void k_feats(const int* x, const int* char_x,
                        const float* wemb, const float* cemb,
                        const float* W2, const float* b2,
                        const float* W3, const float* b3,
                        const float* W4, const float* b4,
                        unsigned short* feats) {
  __shared__ float ce[C_ * CE_];          // [pos][c]
  int bs = blockIdx.x, tid = threadIdx.x;
  for (int i = tid; i < C_ * CE_; i += 128) {
    int pos = i / CE_, c = i % CE_;
    int ch = char_x[bs * C_ + pos];
    ce[i] = cemb[ch * CE_ + c];
  }
  if (tid < E_) {
    int w = x[bs];
    feats[bs * KPAD + tid] = f2bf(wemb[w * E_ + tid]);
  }
  if (tid >= 100 && tid < 117)            // zero pad cols 175..191
    feats[bs * KPAD + 75 + tid] = 0;
  __syncthreads();
  if (tid < 75) {
    int f, wdt; const float* W; float bias;
    if (tid < 25)      { f = tid;      wdt = 2; W = W2 + f * CE_ * 2; bias = b2[f]; }
    else if (tid < 50) { f = tid - 25; wdt = 3; W = W3 + f * CE_ * 3; bias = b3[f]; }
    else               { f = tid - 50; wdt = 4; W = W4 + f * CE_ * 4; bias = b4[f]; }
    float best = -1e30f;
    for (int p = 0; p <= C_ - wdt; ++p) {
      float acc = 0.f;
      for (int c = 0; c < CE_; ++c)
        for (int q = 0; q < wdt; ++q)
          acc += ce[(p + q) * CE_ + c] * W[c * wdt + q];
      best = fmaxf(best, acc);
    }
    feats[bs * KPAD + E_ + tid] = f2bf(fmaxf(best + bias, 0.f));
  }
}

// ---------------- input projection GEMM: pre[dir][s][b][g] = feats @ Wih^T + bias ----------------
// M = 65536 (b*S+s), N = 1024 (dir*512+g), K = 192.  bf16 WMMA, f32 acc.
__global__ void k_inproj(const unsigned short* __restrict__ feats,
                         const unsigned short* __restrict__ WihAll,
                         const float* __restrict__ biasF, const float* __restrict__ biasB,
                         float* __restrict__ pre) {
  int w = threadIdx.x >> 5, lane = threadIdx.x & 31;
  int lm = lane & 15, half = lane >> 4;
  int mt = w & 3, nt = w >> 2;
  int m0 = blockIdx.x * 64 + mt * 16;
  int n0 = blockIdx.y * 32 + nt * 16;
  const unsigned short* arow = feats + (m0 + lm) * KPAD;
  const unsigned short* brow = WihAll + (n0 + lm) * KPAD;
  v8f acc = {0.f, 0.f, 0.f, 0.f, 0.f, 0.f, 0.f, 0.f};
  for (int kc = 0; kc < 6; ++kc) {
    FragB16 a, b;
#pragma unroll
    for (int j = 0; j < 8; ++j) {
      int ka = kc * 32 + ((j < 4) ? (8 * half + 2 * j) : (16 + 8 * half + 2 * j - 8));
      a.d[j] = *(const unsigned int*)(arow + ka);
      int kb = kc * 32 + ((j < 4) ? (8 * half + 2 * j) : (16 + 8 * half + 2 * j - 8));
      b.d[j] = *(const unsigned int*)(brow + kb);
    }
    acc = __builtin_amdgcn_wmma_f32_16x16x32_bf16(false, a.v, false, b.v,
                                                  (short)0, acc, false, false);
  }
  int n = n0 + lm;
  float bias = (n < 512) ? biasF[n] : biasB[n - 512];
  int dir = n >> 9, g = n & 511;
#pragma unroll
  for (int r = 0; r < 8; ++r) {
    int mm = m0 + r + 8 * half;
    int bb = mm >> 9, ss = mm & 511;
    pre[(((size_t)(dir * 512 + ss) * 128 + bb) << 9) + g] = acc[r] + bias;
  }
}

// ---------------- BiLSTM scan: grid (MB_=8 batch-blocks, 2 dirs), 1024 thr = 32 waves ----------------
// Recurrence is independent per batch row: each block owns 16 rows (one WMMA m-tile).
// Per step: each wave computes one 16x16 gate tile (4 wmma, W_hh resident in VGPRs),
// pre[s] loaded into registers BEFORE the GEMM to overlap with WMMA; next step's pre prefetched.
__global__ void __launch_bounds__(1024) k_lstm(const unsigned short* __restrict__ WhhAll,
                                               const float* __restrict__ pre,
                                               float* __restrict__ h_out) {
  __shared__ float gates[16 * G4H];                                    // 32 KB
  __shared__ __attribute__((aligned(16))) unsigned short hbuf[16 * H_]; // 4 KB bf16
  int mb = blockIdx.x;           // batch rows [mb*16, mb*16+16)
  int dir = blockIdx.y;
  int tid = threadIdx.x;
  int w = tid >> 5, lane = tid & 31, lm = lane & 15, half = lane >> 4;

  // W_hh B-fragments resident in VGPRs: wave w owns gate columns [16w,16w+16)
  FragB16 Bfr[4];
  {
    const unsigned short* wrow = WhhAll + (size_t)(dir * G4H + w * 16 + lm) * H_;
#pragma unroll
    for (int kc = 0; kc < 4; ++kc)
#pragma unroll
      for (int j = 0; j < 8; ++j)
        Bfr[kc].d[j] = *(const unsigned int*)(wrow + kc * 32 + ((j < 4) ? (8 * half + 2 * j) : (16 + 8 * half + 2 * j - 8)));
  }
  for (int i = tid; i < 16 * H_; i += 1024) hbuf[i] = 0;
  float creg[2] = {0.f, 0.f};
  __syncthreads();

  for (int t = 0; t < S_; ++t) {
    int s = dir ? (S_ - 1 - t) : t;
    const float* pr = pre + ((size_t)(dir * 512 + s) * 128 + mb * 16) * G4H;
    // ---- issue pre[s] loads early (overlap with WMMA below) ----
    float pgi[2], pgf[2], pgg[2], pgo[2];
#pragma unroll
    for (int r = 0; r < 2; ++r) {
      int idx = r * 1024 + tid;            // = l*128 + hu, l in [0,16)
      int l = idx >> 7, hu = idx & 127;
      int base = l * G4H + hu;
      pgi[r] = pr[base];
      pgf[r] = pr[base + 128];
      pgg[r] = pr[base + 256];
      pgo[r] = pr[base + 384];
    }
    // prefetch next step's pre lines into cache
    if (t + 1 < S_) {
      int sn = dir ? (s - 1) : (s + 1);
      const float* prn = pre + ((size_t)(dir * 512 + sn) * 128 + mb * 16) * G4H;
#pragma unroll
      for (int r = 0; r < 2; ++r) {
        int idx = r * 1024 + tid;
        int l = idx >> 7, hu = idx & 127;
        int base = l * G4H + hu;
        __builtin_prefetch(prn + base, 0, 1);
        __builtin_prefetch(prn + base + 128, 0, 1);
        __builtin_prefetch(prn + base + 256, 0, 1);
        __builtin_prefetch(prn + base + 384, 0, 1);
      }
    }
    // ---- recurrent GEMM: one 16x16 tile per wave, K=128 in 4 chunks ----
    {
      v8f acc = {0.f, 0.f, 0.f, 0.f, 0.f, 0.f, 0.f, 0.f};
      const unsigned short* hrow = hbuf + lm * H_;
#pragma unroll
      for (int kc = 0; kc < 4; ++kc) {
        FragB16 a;
#pragma unroll
        for (int j = 0; j < 8; ++j) {
          int ka = kc * 32 + ((j < 4) ? (8 * half + 2 * j) : (16 + 8 * half + 2 * j - 8));
          a.d[j] = *(const unsigned int*)(hrow + ka);
        }
        acc = __builtin_amdgcn_wmma_f32_16x16x32_bf16(false, a.v, false, Bfr[kc].v,
                                                      (short)0, acc, false, false);
      }
      int g = w * 16 + lm;
#pragma unroll
      for (int r = 0; r < 8; ++r)
        gates[(r + 8 * half) * G4H + g] = acc[r];
    }
    __syncthreads();
    // ---- gate nonlinearities + state update (2 elements/thread, coalesced over hu) ----
    float* ho = h_out + ((size_t)(dir * 512 + s) * 128 + mb * 16) * H_;
#pragma unroll
    for (int r = 0; r < 2; ++r) {
      int idx = r * 1024 + tid;            // = l*128 + hu
      int l = idx >> 7, hu = idx & 127;
      int base = l * G4H + hu;
      float gi = gates[base]       + pgi[r];
      float gf = gates[base + 128] + pgf[r];
      float gg = gates[base + 256] + pgg[r];
      float go = gates[base + 384] + pgo[r];
      float c = sigf(gf) * creg[r] + sigf(gi) * tanhf(gg);
      creg[r] = c;
      float h = sigf(go) * tanhf(c);
      hbuf[idx] = f2bf(h);
      ho[idx] = h;
    }
    __syncthreads();
  }
}

// ---------------- emissions: em[b][s][t] = [h_f,h_b] . h2t_W[t] + h2t_b[t] ----------------
__global__ void k_emis(const float* __restrict__ h_out, const float* __restrict__ W,
                       const float* __restrict__ bvec, float* __restrict__ em) {
  __shared__ float hsh[8][2 * H_];
  int w = threadIdx.x >> 5, lane = threadIdx.x & 31;
  int bs = blockIdx.x * 8 + w;
  int b = bs >> 9, s = bs & 511;
  const float* hf = h_out + ((size_t)(0 * 512 + s) * B_ + b) * H_;
  const float* hb = h_out + ((size_t)(512 + s) * B_ + b) * H_;
  for (int k = lane; k < H_; k += 32) {
    hsh[w][k]      = hf[k];
    hsh[w][H_ + k] = hb[k];
  }
  __syncthreads();
  if (lane < T_) {
    const float* wr = W + lane * (2 * H_);
    float acc = bvec[lane];
    for (int k = 0; k < 2 * H_; ++k) acc += hsh[w][k] * wr[k];
    em[(size_t)bs * T_ + lane] = acc;
  }
}

// ---------------- CRF NLL: one wave per batch row; atomicAdd partial into out ----------------
__global__ void k_crf(const float* __restrict__ em, const int* __restrict__ tags,
                      const unsigned char* __restrict__ mask,
                      const float* __restrict__ start, const float* __restrict__ endv,
                      const float* __restrict__ trans, float* out) {
  __shared__ float tr[T_ * T_];
  __shared__ float alpha[T_];
  int b = blockIdx.x, lane = threadIdx.x;
  for (int i = lane; i < T_ * T_; i += 32) tr[i] = trans[i];
  const float* eb = em + (size_t)b * S_ * T_;
  if (lane < T_) alpha[lane] = start[lane] + eb[lane];
  __syncthreads();
  for (int s = 1; s < S_; ++s) {
    float nv = 0.f;
    if (lane < T_) {
      float m = -1e30f;
      for (int tt = 0; tt < T_; ++tt) m = fmaxf(m, alpha[tt] + tr[tt * T_ + lane]);
      float sum = 0.f;
      for (int tt = 0; tt < T_; ++tt) sum += __expf(alpha[tt] + tr[tt * T_ + lane] - m);
      nv = eb[s * T_ + lane] + m + __logf(sum);
    }
    unsigned char mk = mask[b * S_ + s];
    __syncthreads();
    if (lane < T_ && mk) alpha[lane] = nv;
    __syncthreads();
  }
  if (lane == 0) {
    float m = -1e30f;
    for (int tt = 0; tt < T_; ++tt) m = fmaxf(m, alpha[tt] + endv[tt]);
    float sum = 0.f;
    for (int tt = 0; tt < T_; ++tt) sum += __expf(alpha[tt] + endv[tt] - m);
    float logZ = m + __logf(sum);
    const int* tg = tags + b * S_;
    const unsigned char* mrow = mask + b * S_;
    int t0 = tg[0];
    float sc = start[t0] + eb[t0];
    int cnt = mrow[0] ? 1 : 0;
    for (int s = 1; s < S_; ++s) {
      if (mrow[s]) { sc += tr[tg[s - 1] * T_ + tg[s]] + eb[s * T_ + tg[s]]; ++cnt; }
    }
    sc += endv[tg[cnt - 1]];
    atomicAdd(out, logZ - sc);
  }
}

extern "C" void kernel_launch(void* const* d_in, const int* in_sizes, int n_in,
                              void* d_out, int out_size, void* d_ws, size_t ws_size,
                              hipStream_t stream) {
  const int*   x       = (const int*)d_in[0];
  const int*   char_x  = (const int*)d_in[1];
  const int*   tags    = (const int*)d_in[2];
  const unsigned char* mask = (const unsigned char*)d_in[3];
  const float* wemb    = (const float*)d_in[4];
  const float* cemb    = (const float*)d_in[5];
  const float* c2W = (const float*)d_in[6],  *c2b = (const float*)d_in[7];
  const float* c3W = (const float*)d_in[8],  *c3b = (const float*)d_in[9];
  const float* c4W = (const float*)d_in[10], *c4b = (const float*)d_in[11];
  const float* Wihf = (const float*)d_in[12], *Whhf = (const float*)d_in[13], *bf = (const float*)d_in[14];
  const float* Wihb = (const float*)d_in[15], *Whhb = (const float*)d_in[16], *bb = (const float*)d_in[17];
  const float* h2tW = (const float*)d_in[18], *h2tb = (const float*)d_in[19];
  const float* crfS = (const float*)d_in[20], *crfE = (const float*)d_in[21], *crfT = (const float*)d_in[22];
  float* out = (float*)d_out;

  char* ws = (char*)d_ws;
  size_t off = 0;
  unsigned short* feats  = (unsigned short*)(ws + off); off += (size_t)B_ * S_ * KPAD * 2;        // 25.2 MB
  unsigned short* WihAll = (unsigned short*)(ws + off); off += (size_t)1024 * KPAD * 2;           // 0.4 MB
  unsigned short* WhhAll = (unsigned short*)(ws + off); off += (size_t)2 * G4H * H_ * 2;          // 0.26 MB
  off = (off + 255) & ~(size_t)255;
  float* pre   = (float*)(ws + off); off += (size_t)2 * S_ * B_ * G4H * 4;                        // 268 MB
  float* h_out = (float*)(ws + off); off += (size_t)2 * S_ * B_ * H_ * 4;                         // 67 MB
  float* em    = (float*)(ws + off); off += (size_t)B_ * S_ * T_ * 4;                             // 4.5 MB

  k_zero<<<1, 32, 0, stream>>>(out);
  k_prep<<<(1024 * KPAD + 255) / 256, 256, 0, stream>>>(Wihf, Wihb, Whhf, Whhb, WihAll, WhhAll);
  k_feats<<<B_ * S_, 128, 0, stream>>>(x, char_x, wemb, cemb, c2W, c2b, c3W, c3b, c4W, c4b, feats);
  k_inproj<<<dim3(B_ * S_ / 64, 1024 / 32), 256, 0, stream>>>(feats, WihAll, bf, bb, pre);
  k_lstm<<<dim3(MB_, 2), 1024, 0, stream>>>(WhhAll, pre, h_out);
  k_emis<<<B_ * S_ / 8, 256, 0, stream>>>(h_out, h2tW, h2tb, em);
  k_crf<<<B_, 32, 0, stream>>>(em, tags, mask, crfS, crfE, crfT, out);
}
